// MultiHeadAttention_34789235098192
// MI455X (gfx1250) — matrix operand order repaired and checked by reading, not compiled, not run on hardware
//
#include <hip/hip_runtime.h>
#include <hip/hip_bf16.h>

// ---------------------------------------------------------------------------
// MultiHeadAttention (B=4, S=2048, D=1024, E=head*dim=1024) for gfx1250.
// All GEMMs are "NT": C[m,n] = scale * sum_k A[m,k] * B[n,k], bf16 inputs,
// f32 accumulation via v_wmma_f32_16x16x32_bf16, LDS double-buffered with
// global_load_async_to_lds_b128 (ASYNCcnt) staging.
//
// Pipeline:
//   1. cvt  : query/key/value f32 -> bf16
//   2. tcvt : Wq,Wk,Wv -> WqT,WkT,WvT (bf16 [E,D]); Wo -> WoT (bf16 [D,E])
//   3. Q  = Xq (S,D) x WqT (E,D)          -> bf16 [S,E]
//      K  = Xk x WkT                      -> bf16 [S,E]
//      VT = WvT (E,D) x Xv (S,D)          -> bf16 [E,S]  (projection, pre-transposed)
//   4. scores = Q (S,E) x K (S,E) * 1/8   -> f32 [S,S]
//   5. softmax over axis=1 (query axis)   -> bf16 attn [S,S]
//   6. head = attn (S,S) x VT (E,S)       -> bf16 [S,E]
//   7. out  = head (S,E) x WoT (D,E)      -> f32 [S,D] (d_out)
// ---------------------------------------------------------------------------

typedef __bf16 bf16_t;
typedef __attribute__((ext_vector_type(16))) __bf16 v16bf;
typedef __attribute__((ext_vector_type(8)))  __bf16 v8bf;
typedef __attribute__((ext_vector_type(8)))  float  v8f;
typedef __attribute__((ext_vector_type(4)))  float  v4f;

union V16U { v16bf v; v8bf h[2]; };

__device__ __forceinline__ void store_c(float*  p, float v) { *p = v; }
__device__ __forceinline__ void store_c(bf16_t* p, float v) { *p = (bf16_t)v; }

// Async 16-byte global -> LDS copy, tracked by ASYNCcnt (cdna5 §10 async ops).
// lds_off is the wave-relative LDS byte offset (ptrtoint of an AS(3) pointer).
__device__ __forceinline__ void async_copy_b128(unsigned lds_off,
                                                const bf16_t* gptr) {
    asm volatile("global_load_async_to_lds_b128 %0, %1, off"
                 :: "v"(lds_off), "v"(gptr)
                 : "memory");
}

__device__ __forceinline__ void wait_async0() {
    asm volatile("s_wait_asynccnt 0x0" ::: "memory");
}

__device__ __forceinline__ unsigned lds_off_of(const bf16_t* p) {
    // p points into __shared__; ptrtoint of an AS(3) pointer is the LDS offset.
    return (unsigned)(unsigned long long)p;
}

// ---------------------------------------------------------------------------
// NT GEMM: C[m,n] = scale * sum_k A[m,k] * B[n,k]
// A: [M,K] bf16 row-major, B: [N,K] bf16 row-major, C: [M,N] (f32 or bf16).
// Block: 256 threads = 8 waves (4 x 2). Block tile 128x128, wave tile 32x64,
// K-step 32, LDS double-buffered via async copies.
// grid = (N/128, M/128, batch). M,N % 128 == 0, K % 32 == 0.
// ---------------------------------------------------------------------------
#define LDSPAD 40   // padded row stride in bf16 elements (80 B, 16B-aligned)

template <typename OutT>
__global__ __launch_bounds__(256) void gemm_nt_wmma(
    const bf16_t* __restrict__ A, long long sA,
    const bf16_t* __restrict__ Bm, long long sB,
    OutT* __restrict__ C, long long sC,
    int M, int N, int K, float scale)
{
    __shared__ bf16_t As[2][128][LDSPAD];
    __shared__ bf16_t Bs[2][128][LDSPAD];

    const int batch = blockIdx.z;
    const bf16_t* Ab = A  + (long long)batch * sA + (size_t)(blockIdx.y * 128) * K;
    const bf16_t* Bb = Bm + (long long)batch * sB + (size_t)(blockIdx.x * 128) * K;
    OutT*         Cb = C  + (long long)batch * sC;

    const int tid  = threadIdx.x;
    const int wave = tid >> 5;
    const int lane = tid & 31;
    const int hi   = lane >> 4;       // half-wave select
    const int l16  = lane & 15;
    const int wm   = (wave >> 1) * 32;   // wave row offset within 128-tile
    const int wn   = (wave & 1) * 64;    // wave col offset within 128-tile

    // Copy-chunk assignment: each tile is 128 rows x 4 sixteen-byte chunks
    // (512 chunks); each of 256 threads moves chunk tid and tid+256,
    // for A and for B (4 async b128 issues per thread per K-step).
    const int r0 = tid >> 2,          p0 = tid & 3;
    const int r1 = (tid + 256) >> 2,  p1 = (tid + 256) & 3;

    v8f acc[2][4];
#pragma unroll
    for (int f = 0; f < 2; ++f)
#pragma unroll
        for (int t = 0; t < 4; ++t) acc[f][t] = (v8f)0.0f;

    auto issue = [&](int kk, int buf) {
        async_copy_b128(lds_off_of(&As[buf][r0][p0 * 8]),
                        Ab + (size_t)r0 * K + kk + p0 * 8);
        async_copy_b128(lds_off_of(&As[buf][r1][p1 * 8]),
                        Ab + (size_t)r1 * K + kk + p1 * 8);
        async_copy_b128(lds_off_of(&Bs[buf][r0][p0 * 8]),
                        Bb + (size_t)r0 * K + kk + p0 * 8);
        async_copy_b128(lds_off_of(&Bs[buf][r1][p1 * 8]),
                        Bb + (size_t)r1 * K + kk + p1 * 8);
    };

    // Prologue: fill buffer 0.
    issue(0, 0);
    wait_async0();
    __syncthreads();

    int buf = 0;
    for (int kk = 0; kk < K; kk += 32) {
        if (kk + 32 < K) issue(kk + 32, buf ^ 1);

        // A fragments (16x32 bf16 A-layout: M = lane&15; lanes 0-15 take
        // K {0..7,16..23}, lanes 16-31 take K {8..15,24..31}).
        V16U a[2];
#pragma unroll
        for (int f = 0; f < 2; ++f) {
            const bf16_t* ar = &As[buf][wm + f * 16 + l16][0];
            a[f].h[0] = *(const v8bf*)(ar + hi * 8);
            a[f].h[1] = *(const v8bf*)(ar + 16 + hi * 8);
        }
        // B fragments (32x16 bf16 B-layout: column = lane&15; lanes 0-15
        // hold K {0..7,16..23}, lanes 16-31 hold K {8..15,24..31}).
#pragma unroll
        for (int t = 0; t < 4; ++t) {
            const bf16_t* br = &Bs[buf][wn + t * 16 + l16][0];
            V16U b;
            b.h[0] = *(const v8bf*)(br + hi * 8);
            b.h[1] = *(const v8bf*)(br + 16 + hi * 8);
            acc[0][t] = __builtin_amdgcn_wmma_f32_16x16x32_bf16(
                false, a[0].v, false, b.v, (short)0, acc[0][t], false, false);
            acc[1][t] = __builtin_amdgcn_wmma_f32_16x16x32_bf16(
                false, a[1].v, false, b.v, (short)0, acc[1][t], false, false);
        }

        wait_async0();
        __syncthreads();
        buf ^= 1;
    }

    // C/D f32 layout: VGPR v holds row (v + 8*hi), column = lane&15.
#pragma unroll
    for (int f = 0; f < 2; ++f) {
#pragma unroll
        for (int t = 0; t < 4; ++t) {
            const int col = blockIdx.x * 128 + wn + t * 16 + l16;
#pragma unroll
            for (int v = 0; v < 8; ++v) {
                const int row = blockIdx.y * 128 + wm + f * 16 + v + hi * 8;
                store_c(Cb + (size_t)row * N + col, acc[f][t][v] * scale);
            }
        }
    }
}

// f32 -> bf16 elementwise (n % 4 == 0), vectorized 4-wide.
__global__ __launch_bounds__(256) void cvt_bf16_kernel(
    const float* __restrict__ src, bf16_t* __restrict__ dst, long long n4)
{
    long long i = (long long)blockIdx.x * blockDim.x + threadIdx.x;
    if (i >= n4) return;
    v4f x = *(const v4f*)(src + 4 * i);
    bf16_t* d = dst + 4 * i;
    d[0] = (bf16_t)x.x; d[1] = (bf16_t)x.y;
    d[2] = (bf16_t)x.z; d[3] = (bf16_t)x.w;
}

// f32 [R,C] -> bf16 [C,R] LDS-tiled transpose. R,C % 32 == 0. block (32,8).
__global__ __launch_bounds__(256) void transpose_cvt_kernel(
    const float* __restrict__ src, bf16_t* __restrict__ dst, int R, int C)
{
    __shared__ bf16_t tile[32][33];
    const int c0 = blockIdx.x * 32;
    const int r0 = blockIdx.y * 32;
#pragma unroll
    for (int i = threadIdx.y; i < 32; i += 8)
        tile[i][threadIdx.x] =
            (bf16_t)src[(size_t)(r0 + i) * C + c0 + threadIdx.x];
    __syncthreads();
#pragma unroll
    for (int i = threadIdx.y; i < 32; i += 8)
        dst[(size_t)(c0 + i) * R + r0 + threadIdx.x] = tile[threadIdx.x][i];
}

// Softmax over the QUERY axis (axis=1 quirk of the reference):
// for each (batch, column k): attn[q,k] = softmax_q(scores[q,k]).
// One thread per column; accesses are coalesced across threads at fixed q.
// Online max+sum in a single pass, then one normalize+store pass.
__global__ __launch_bounds__(256) void softmax_q_kernel(
    const float* __restrict__ scores, bf16_t* __restrict__ attn, int S)
{
    const float* sc = scores + (size_t)blockIdx.y * S * S;
    bf16_t*      at = attn   + (size_t)blockIdx.y * S * S;
    const int k = blockIdx.x * 256 + threadIdx.x;

    float mx = -3.402823466e+38f;
    float sum = 0.0f;
    for (int q = 0; q < S; ++q) {
        const float x = sc[(size_t)q * S + k];
        const float m2 = fmaxf(mx, x);
        sum = sum * __expf(mx - m2) + __expf(x - m2);
        mx = m2;
    }
    const float inv = 1.0f / sum;
    for (int q = 0; q < S; ++q)
        at[(size_t)q * S + k] =
            (bf16_t)(__expf(sc[(size_t)q * S + k] - mx) * inv);
}

// ---------------------------------------------------------------------------

extern "C" void kernel_launch(void* const* d_in, const int* in_sizes, int n_in,
                              void* d_out, int out_size, void* d_ws, size_t ws_size,
                              hipStream_t stream)
{
    constexpr int  B = 4, S = 2048, D = 1024, E = 1024;
    constexpr long long BS = (long long)B * S;

    const float* q  = (const float*)d_in[0];
    const float* kk = (const float*)d_in[1];
    const float* vv = (const float*)d_in[2];
    const float* Wq = (const float*)d_in[3];
    const float* Wk = (const float*)d_in[4];
    const float* Wv = (const float*)d_in[5];
    const float* Wo = (const float*)d_in[6];
    float* out = (float*)d_out;

    char* ws = (char*)d_ws;
    size_t off = 0;
    auto carve = [&](size_t bytes) -> char* {
        char* p = ws + off;
        off += (bytes + 255) & ~(size_t)255;
        return p;
    };

    bf16_t* Xq  = (bf16_t*)carve(BS * D * 2);            // bf16 [B,S,D]
    bf16_t* Xk  = (bf16_t*)carve(BS * D * 2);
    bf16_t* Xv  = (bf16_t*)carve(BS * D * 2);
    bf16_t* WqT = (bf16_t*)carve((size_t)E * D * 2);     // bf16 [E,D]
    bf16_t* WkT = (bf16_t*)carve((size_t)E * D * 2);
    bf16_t* WvT = (bf16_t*)carve((size_t)E * D * 2);
    bf16_t* WoT = (bf16_t*)carve((size_t)D * E * 2);     // bf16 [D,E]
    bf16_t* Qp  = (bf16_t*)carve(BS * E * 2);            // bf16 [B,S,E]
    bf16_t* Kp  = (bf16_t*)carve(BS * E * 2);
    bf16_t* VT  = (bf16_t*)carve((size_t)B * E * S * 2); // bf16 [B,E,S]
    float*  SC  = (float*) carve((size_t)B * S * S * 4); // f32  [B,S,S]
    bf16_t* AT  = (bf16_t*)carve((size_t)B * S * S * 2); // bf16 [B,S,S]
    bf16_t* HD  = (bf16_t*)carve(BS * E * 2);            // bf16 [B,S,E]

    // 1. convert activations to bf16
    {
        const long long n4 = BS * D / 4;
        const int blocks = (int)((n4 + 255) / 256);
        cvt_bf16_kernel<<<blocks, 256, 0, stream>>>(q,  Xq, n4);
        cvt_bf16_kernel<<<blocks, 256, 0, stream>>>(kk, Xk, n4);
        cvt_bf16_kernel<<<blocks, 256, 0, stream>>>(vv, Xv, n4);
    }
    // 2. transpose-convert weights
    transpose_cvt_kernel<<<dim3(E / 32, D / 32), dim3(32, 8), 0, stream>>>(Wq, WqT, D, E);
    transpose_cvt_kernel<<<dim3(E / 32, D / 32), dim3(32, 8), 0, stream>>>(Wk, WkT, D, E);
    transpose_cvt_kernel<<<dim3(E / 32, D / 32), dim3(32, 8), 0, stream>>>(Wv, WvT, D, E);
    transpose_cvt_kernel<<<dim3(D / 32, E / 32), dim3(32, 8), 0, stream>>>(Wo, WoT, E, D);

    // 3. projections
    gemm_nt_wmma<bf16_t><<<dim3(E / 128, S / 128, B), 256, 0, stream>>>(
        Xq, (long long)S * D, WqT, 0, Qp, (long long)S * E, S, E, D, 1.0f);
    gemm_nt_wmma<bf16_t><<<dim3(E / 128, S / 128, B), 256, 0, stream>>>(
        Xk, (long long)S * D, WkT, 0, Kp, (long long)S * E, S, E, D, 1.0f);
    // VT[b] (E,S) = WvT (E,D) x Xv[b] (S,D)
    gemm_nt_wmma<bf16_t><<<dim3(S / 128, E / 128, B), 256, 0, stream>>>(
        WvT, 0, Xv, (long long)S * D, VT, (long long)E * S, E, S, D, 1.0f);

    // 4. scores = Q x K^T * (1/sqrt(64))
    gemm_nt_wmma<float><<<dim3(S / 128, S / 128, B), 256, 0, stream>>>(
        Qp, (long long)S * E, Kp, (long long)S * E, SC, (long long)S * S,
        S, S, E, 0.125f);

    // 5. softmax over query axis
    softmax_q_kernel<<<dim3(S / 256, B), 256, 0, stream>>>(SC, AT, S);

    // 6. head = attn x V
    gemm_nt_wmma<bf16_t><<<dim3(E / 128, S / 128, B), 256, 0, stream>>>(
        AT, (long long)S * S, VT, (long long)E * S, HD, (long long)S * E,
        S, E, S, 1.0f);

    // 7. out = head x Wo
    gemm_nt_wmma<float><<<dim3(D / 128, S / 128, B), 256, 0, stream>>>(
        HD, (long long)S * E, WoT, 0, out, (long long)S * D, S, D, E, 1.0f);
}
